// PretrainGATModel_45981919871352
// MI455X (gfx1250) — compile-verified, matrix operand order checked by reading, not run on hardware
//
#include <hip/hip_runtime.h>

#define NN 50000
#define NE 800000
#define INF_DIM 256
#define HID 64
#define HEADS 8
#define FDIM (HEADS * HID) /* 512 */

typedef __bf16 v16bf __attribute__((ext_vector_type(16)));
typedef float v8f __attribute__((ext_vector_type(8)));
typedef unsigned short u16x8 __attribute__((ext_vector_type(8)));

// ---------- helpers ----------
static __device__ __forceinline__ unsigned short f2bf(float f) {
    unsigned u = __float_as_uint(f);
    unsigned r = u + 0x7FFFu + ((u >> 16) & 1u); // round-to-nearest-even
    return (unsigned short)(r >> 16);
}
static __device__ __forceinline__ float bf2f(unsigned short h) {
    return __uint_as_float(((unsigned)h) << 16);
}
// order-preserving float <-> uint for atomicMax
static __device__ __forceinline__ unsigned fenc(float f) {
    unsigned u = __float_as_uint(f);
    return (u & 0x80000000u) ? ~u : (u | 0x80000000u);
}
static __device__ __forceinline__ float fdec(unsigned u) {
    u = (u & 0x80000000u) ? (u & 0x7FFFFFFFu) : ~u;
    return __uint_as_float(u);
}
static __device__ __forceinline__ v16bf cat16(u16x8 lo, u16x8 hi) {
    return __builtin_bit_cast(
        v16bf, __builtin_shufflevector(lo, hi, 0, 1, 2, 3, 4, 5, 6, 7, 8, 9, 10, 11, 12, 13, 14, 15));
}

// ---------- split fp32 -> bf16 hi/lo ----------
__global__ void split_bf16_kernel(const float* __restrict__ in,
                                  unsigned short* __restrict__ hi,
                                  unsigned short* __restrict__ lo, int n) {
    int i = blockIdx.x * blockDim.x + threadIdx.x;
    if (i < n) {
        float f = in[i];
        unsigned short h = f2bf(f);
        hi[i] = h;
        lo[i] = f2bf(f - bf2f(h));
    }
}

__global__ void fill_f32_kernel(float* __restrict__ p, float v, int n) {
    int i = blockIdx.x * blockDim.x + threadIdx.x;
    if (i < n) p[i] = v;
}
__global__ void fill_u32_kernel(unsigned* __restrict__ p, unsigned v, int n) {
    int i = blockIdx.x * blockDim.x + threadIdx.x;
    if (i < n) p[i] = v;
}

// ---------- WMMA GEMM: C[M,Nout] = A[M,K] * B[Nout,K]^T (+bias, relu) ----------
// bf16 hi/lo splits, 3-product compensation (~fp32 accuracy).
// Register blocking: each wave computes a 16M x 64N strip (4 WMMA tiles),
// holding one A hi/lo pair in VGPRs across all 4 B tiles per K-step.
__global__ __launch_bounds__(256) void gemm_bf16x3_wmma(
    const unsigned short* __restrict__ Ahi, const unsigned short* __restrict__ Alo,
    const unsigned short* __restrict__ Bhi, const unsigned short* __restrict__ Blo,
    float* __restrict__ C, int M, int K, int Nout,
    const float* __restrict__ bias, int do_relu) {
    int wave = (int)((blockIdx.x * blockDim.x + threadIdx.x) >> 5);
    int lane = threadIdx.x & 31;
    int tiles_n = Nout >> 6; // 64-wide N blocks
    int ntiles = (M >> 4) * tiles_n;
    if (wave >= ntiles) return; // wave-uniform exit; EXEC all-1 inside
    int tm = wave / tiles_n;
    int tnb = wave - tm * tiles_n;
    int ra = tm * 16 + (lane & 15); // A row for this lane
    int kh = (lane >> 4) << 3;      // lanes 16..31 take K offset +8 (ISA 16-bit A/B layout)
    const unsigned short* arh = Ahi + (size_t)ra * K + kh;
    const unsigned short* arl = Alo + (size_t)ra * K + kh;
    size_t boff = (size_t)(tnb * 64 + (lane & 15)) * K + kh; // base B row (= weight row)
    v8f acc[4] = {};
    for (int k0 = 0; k0 < K; k0 += 32) {
        u16x8 a0 = *(const u16x8*)(arh + k0);
        u16x8 a1 = *(const u16x8*)(arh + k0 + 16);
        u16x8 l0 = *(const u16x8*)(arl + k0);
        u16x8 l1 = *(const u16x8*)(arl + k0 + 16);
        v16bf ah = cat16(a0, a1);
        v16bf al = cat16(l0, l1);
#pragma unroll
        for (int t = 0; t < 4; ++t) {
            const unsigned short* bhp = Bhi + boff + (size_t)(t * 16) * K + k0;
            const unsigned short* blp = Blo + boff + (size_t)(t * 16) * K + k0;
            u16x8 b0 = *(const u16x8*)(bhp);
            u16x8 b1 = *(const u16x8*)(bhp + 16);
            u16x8 c0 = *(const u16x8*)(blp);
            u16x8 c1 = *(const u16x8*)(blp + 16);
            v16bf bh = cat16(b0, b1);
            v16bf bl = cat16(c0, c1);
            acc[t] = __builtin_amdgcn_wmma_f32_16x16x32_bf16(false, ah, false, bh, (short)0, acc[t], false, false);
            acc[t] = __builtin_amdgcn_wmma_f32_16x16x32_bf16(false, al, false, bh, (short)0, acc[t], false, false);
            acc[t] = __builtin_amdgcn_wmma_f32_16x16x32_bf16(false, ah, false, bl, (short)0, acc[t], false, false);
        }
    }
    int mbase = tm * 16 + ((lane >> 4) << 3);
#pragma unroll
    for (int t = 0; t < 4; ++t) {
        int n = tnb * 64 + t * 16 + (lane & 15);
        float bv = bias ? bias[n] : 0.0f;
#pragma unroll
        for (int v = 0; v < 8; ++v) {
            float val = acc[t][v] + bv;
            if (do_relu) val = fmaxf(val, 0.0f);
            C[(size_t)(mbase + v) * Nout + n] = val;
        }
    }
}

// ---------- per-node attention logits: s[n,h] = dot(h[n,h,:], att[h,:]) ----------
__global__ void logits_kernel(const float* __restrict__ h,
                              const float* __restrict__ att_src,
                              const float* __restrict__ att_dst,
                              float* __restrict__ s1, float* __restrict__ s2,
                              int n_nodes) {
    int wave = (int)((blockIdx.x * blockDim.x + threadIdx.x) >> 5);
    int lane = threadIdx.x & 31;
    if (wave >= n_nodes) return;
    int head = lane >> 2;     // 8 heads, 4 lanes each
    int c0 = (lane & 3) * 16; // 16 channels per lane
    const float* hp = h + (size_t)wave * FDIM + head * HID + c0;
    const float* ap = att_src + head * HID + c0;
    const float* dp = att_dst + head * HID + c0;
    float p1 = 0.f, p2 = 0.f;
#pragma unroll
    for (int i = 0; i < 16; ++i) {
        float v = hp[i];
        p1 += v * ap[i];
        p2 += v * dp[i];
    }
    p1 += __shfl_xor(p1, 1, 32); p1 += __shfl_xor(p1, 2, 32);
    p2 += __shfl_xor(p2, 1, 32); p2 += __shfl_xor(p2, 2, 32);
    if ((lane & 3) == 0) {
        s1[wave * HEADS + head] = p1;
        s2[wave * HEADS + head] = p2;
    }
}

// ---------- edge pass 1: alpha = leakyrelu(s1[dst]+s2[src]); seg max over dst ----------
__global__ void edge_alpha_kernel(const int* __restrict__ ei,
                                  const float* __restrict__ s1,
                                  const float* __restrict__ s2,
                                  float* __restrict__ alpha,
                                  unsigned* __restrict__ segmax, int n_edges) {
    int idx = blockIdx.x * blockDim.x + threadIdx.x;
    if (idx >= n_edges * HEADS) return;
    int e = idx >> 3, hh = idx & 7;
    int src = ei[e];
    int dst = ei[n_edges + e];
    float a = s1[dst * HEADS + hh] + s2[src * HEADS + hh];
    a = (a > 0.f) ? a : 0.2f * a;
    alpha[idx] = a;
    atomicMax(&segmax[dst * HEADS + hh], fenc(a));
}

// ---------- edge pass 2: e = exp(alpha - max[dst]); denom[dst] += e ----------
__global__ void edge_exp_kernel(const int* __restrict__ ei,
                                const unsigned* __restrict__ segmax,
                                float* __restrict__ alpha,
                                float* __restrict__ denom, int n_edges) {
    int idx = blockIdx.x * blockDim.x + threadIdx.x;
    if (idx >= n_edges * HEADS) return;
    int e = idx >> 3, hh = idx & 7;
    int dst = ei[n_edges + e];
    float m = fdec(segmax[dst * HEADS + hh]);
    float ev = expf(alpha[idx] - m);
    alpha[idx] = ev;
    atomicAdd(&denom[dst * HEADS + hh], ev);
}

// ---------- edge pass 3: acc[dst,h,c] += (e/denom[dst,h]) * h[src,h,c] ----------
// 64 threads per edge, 8 consecutive channels each (same head within a thread).
__global__ void edge_scatter_kernel(const int* __restrict__ ei,
                                    const float* __restrict__ h,
                                    const float* __restrict__ ew,
                                    const float* __restrict__ denom,
                                    float* __restrict__ acc, int n_edges) {
    int idx = blockIdx.x * blockDim.x + threadIdx.x;
    if (idx >= n_edges * 64) return;
    int e = idx >> 6;
    int j = idx & 63;
    int f0 = j * 8;
    int hh = f0 >> 6;
    int src = ei[e];
    int dst = ei[n_edges + e];
    float coeff = ew[e * HEADS + hh] / denom[dst * HEADS + hh];
    const float* hs = h + (size_t)src * FDIM + f0;
    float* op = acc + (size_t)dst * FDIM + f0;
#pragma unroll
    for (int i = 0; i < 8; ++i) atomicAdd(op + i, coeff * hs[i]);
}

// ---------- mean over heads + relu: out[n,c] = relu(mean_h acc[n,h,c]) ----------
__global__ void mean_relu_kernel(const float* __restrict__ acc,
                                 float* __restrict__ out, int n_nodes) {
    int idx = blockIdx.x * blockDim.x + threadIdx.x;
    if (idx >= n_nodes * HID) return;
    int n = idx >> 6, c = idx & 63;
    float s = 0.f;
#pragma unroll
    for (int hh = 0; hh < HEADS; ++hh) s += acc[(size_t)n * FDIM + hh * HID + c];
    out[idx] = fmaxf(s * 0.125f, 0.f);
}

__global__ void mask_out_kernel(const unsigned char* __restrict__ mask,
                                float* __restrict__ out, int n) {
    int i = blockIdx.x * blockDim.x + threadIdx.x;
    if (i < n) out[i] = mask[i] ? 1.0f : 0.0f;
}

// ================= host =================
extern "C" void kernel_launch(void* const* d_in, const int* in_sizes, int n_in,
                              void* d_out, int out_size, void* d_ws, size_t ws_size,
                              hipStream_t stream) {
    (void)in_sizes; (void)n_in; (void)out_size; (void)ws_size;
    const float* x   = (const float*)d_in[0];
    const int*   ei  = (const int*)d_in[1];
    const unsigned char* msk = (const unsigned char*)d_in[2];
    const float* W1  = (const float*)d_in[3];
    const float* as1 = (const float*)d_in[4];
    const float* ad1 = (const float*)d_in[5];
    const float* W2  = (const float*)d_in[6];
    const float* as2 = (const float*)d_in[7];
    const float* ad2 = (const float*)d_in[8];
    const float* Wp1 = (const float*)d_in[9];
    const float* bp1 = (const float*)d_in[10];
    const float* Wp2 = (const float*)d_in[11];
    const float* bp2 = (const float*)d_in[12];
    float* out = (float*)d_out;

    // workspace carve-out (256B aligned)
    char* ws = (char*)d_ws;
    size_t off = 0;
    auto alloc = [&](size_t bytes) -> char* {
        char* p = ws + off;
        off = (off + bytes + 255) & ~(size_t)255;
        return p;
    };
    float*          hbuf  = (float*)alloc((size_t)NN * FDIM * 4);   // GEMM output h [N,512]
    float*          accb  = (float*)alloc((size_t)NN * FDIM * 4);   // message accumulator
    float*          abuf  = (float*)alloc((size_t)NE * HEADS * 4);  // alpha / e per edge
    float*          s1    = (float*)alloc((size_t)NN * HEADS * 4);
    float*          s2    = (float*)alloc((size_t)NN * HEADS * 4);
    float*          denom = (float*)alloc((size_t)NN * HEADS * 4);
    unsigned*       segmx = (unsigned*)alloc((size_t)NN * HEADS * 4);
    float*          hact  = (float*)alloc((size_t)NN * HID * 4);    // layer activation [N,64]
    float*          pact  = (float*)alloc((size_t)NN * HID * 4);    // predictor hidden
    unsigned short* ahi   = (unsigned short*)alloc((size_t)NN * INF_DIM * 2);
    unsigned short* alo   = (unsigned short*)alloc((size_t)NN * INF_DIM * 2);
    unsigned short* whi   = (unsigned short*)alloc((size_t)FDIM * INF_DIM * 2);
    unsigned short* wlo   = (unsigned short*)alloc((size_t)FDIM * INF_DIM * 2);

    auto split = [&](const float* in, unsigned short* hi, unsigned short* lo, int n) {
        split_bf16_kernel<<<(n + 255) / 256, 256, 0, stream>>>(in, hi, lo, n);
    };
    auto gemm = [&](const unsigned short* Ah, const unsigned short* Al,
                    const unsigned short* Bh, const unsigned short* Bl,
                    float* C, int M, int K, int Nout, const float* bias, int relu) {
        int ntiles = (M / 16) * (Nout / 64); // one 16x64 strip per wave
        gemm_bf16x3_wmma<<<(ntiles + 7) / 8, 256, 0, stream>>>(Ah, Al, Bh, Bl, C, M, K, Nout, bias, relu);
    };
    auto run_gat = [&](const float* att_s, const float* att_d) {
        logits_kernel<<<(NN * 32 + 255) / 256, 256, 0, stream>>>(hbuf, att_s, att_d, s1, s2, NN);
        fill_u32_kernel<<<(NN * HEADS + 255) / 256, 256, 0, stream>>>(segmx, 0x007FFFFFu, NN * HEADS); // enc(-inf)
        fill_f32_kernel<<<(NN * HEADS + 255) / 256, 256, 0, stream>>>(denom, 0.f, NN * HEADS);
        fill_f32_kernel<<<(NN * FDIM + 255) / 256, 256, 0, stream>>>(accb, 0.f, NN * FDIM);
        edge_alpha_kernel<<<(NE * HEADS + 255) / 256, 256, 0, stream>>>(ei, s1, s2, abuf, segmx, NE);
        edge_exp_kernel<<<(NE * HEADS + 255) / 256, 256, 0, stream>>>(ei, segmx, abuf, denom, NE);
        edge_scatter_kernel<<<(NE * 64 + 255) / 256, 256, 0, stream>>>(ei, hbuf, abuf, denom, accb, NE);
        mean_relu_kernel<<<(NN * HID + 255) / 256, 256, 0, stream>>>(accb, hact, NN);
    };

    // ---- GAT layer 1 ----
    split(x, ahi, alo, NN * INF_DIM);
    split(W1, whi, wlo, FDIM * INF_DIM);
    gemm(ahi, alo, whi, wlo, hbuf, NN, INF_DIM, FDIM, nullptr, 0);
    run_gat(as1, ad1);

    // ---- GAT layer 2 ----
    split(hact, ahi, alo, NN * HID);
    split(W2, whi, wlo, FDIM * HID);
    gemm(ahi, alo, whi, wlo, hbuf, NN, HID, FDIM, nullptr, 0);
    run_gat(as2, ad2);

    // ---- predictor: relu(h@Wp1^T + b1) @ Wp2^T + b2 ----
    split(hact, ahi, alo, NN * HID);
    split(Wp1, whi, wlo, HID * HID);
    gemm(ahi, alo, whi, wlo, pact, NN, HID, HID, bp1, 1);

    split(pact, ahi, alo, NN * HID);
    split(Wp2, whi, wlo, INF_DIM * HID);
    gemm(ahi, alo, whi, wlo, out, NN, HID, INF_DIM, bp2, 0);

    // ---- mask passthrough (second tuple element) ----
    mask_out_kernel<<<(NN + 255) / 256, 256, 0, stream>>>(msk, out + (size_t)NN * INF_DIM, NN);
}
